// WindowAttention_67765993997103
// MI455X (gfx1250) — matrix operand order repaired and checked by reading, not compiled, not run on hardware
//
#include <hip/hip_runtime.h>
#include <hip/hip_bf16.h>

typedef __attribute__((ext_vector_type(16))) _Float16 v16h;
typedef __attribute__((ext_vector_type(8)))  _Float16 v8h;
typedef __attribute__((ext_vector_type(8)))  float    v8f;

#define DEVINL static __device__ __forceinline__

DEVINL v8f wmma16(v16h a, v16h b, v8f c) {
  // D = A(16x32 f16) * B(32x16 f16) + C(16x16 f32)
  return __builtin_amdgcn_wmma_f32_16x16x32_f16(false, a, false, b, (short)0, c, false, false);
}

// A-operand (16x32 f16) load: lane m=lane&15 holds
// elems 0..15 = K (kbase)+0..15  with kbase = (lane<16 ? 0 : 16)
// Caller passes p = rowBase + kk + (hi?16:0); second half at p+8.
DEVINL v16h load_a16(const _Float16* p) {
  union { v16h v; v8h h2[2]; } u;
  u.h2[0] = *(const v8h*)(p);
  u.h2[1] = *(const v8h*)(p + 8);
  return u.v;
}

// ---------------------------------------------------------------------------
// Kernel: convert fp32 weights to f16 workspace copies.
__global__ void k_cvt(const float* __restrict__ qw, const float* __restrict__ pw,
                      _Float16* __restrict__ qh, _Float16* __restrict__ ph) {
  int idx = blockIdx.x * 256 + threadIdx.x;
  if (idx < 786432) qh[idx] = (_Float16)qw[idx];   // qkv_w: 1536*512
  if (idx < 262144) ph[idx] = (_Float16)pw[idx];   // proj_w: 512*512
}

// ---------------------------------------------------------------------------
// Kernel: CPB MLP table (225 positions x 16 heads), pre-applies 16*sigmoid.
__global__ void k_table(const float* __restrict__ rct, const float* __restrict__ w1,
                        const float* __restrict__ b1, const float* __restrict__ w2,
                        float* __restrict__ table) {
  int idx = blockIdx.x * 256 + threadIdx.x;
  if (idx >= 225 * 16) return;
  int p = idx >> 4, h = idx & 15;
  float t0 = rct[p * 2 + 0], t1 = rct[p * 2 + 1];
  float acc = 0.f;
  for (int j = 0; j < 512; ++j) {
    float hd = fmaf(t0, w1[j * 2 + 0], fmaf(t1, w1[j * 2 + 1], b1[j]));
    hd = fmaxf(hd, 0.f);
    acc = fmaf(hd, w2[h * 512 + j], acc);
  }
  table[p * 16 + h] = 16.f / (1.f + __expf(-acc));
}

// ---------------------------------------------------------------------------
// Kernel: full bias[h][i][j] (128x128 per head) with adapters.
__global__ void k_bias(const int* __restrict__ rpi, const float* __restrict__ table,
                       const float* __restrict__ a1, const float* __restrict__ a2,
                       float* __restrict__ biasb) {
  int idx = blockIdx.x * 256 + threadIdx.x;           // 16*128*128
  int h = idx >> 14;
  int rem = idx & 16383;
  int i = rem >> 7, j = rem & 127;
  int ip = i & 63;
  float t = table[rpi[ip * 64 + (j & 63)] * 16 + h];
  float add = (j < 64) ? 2.f * a1[(h * 64 + ip) * 64 + j]
                       : a2[(h * 64 + ip) * 64 + (j - 64)];
  biasb[idx] = t + add;
}

// ---------------------------------------------------------------------------
// Fused window attention: 1 block per window. 256 threads = 8 wave32s,
// each wave owns a 16-row tile of the 128 concatenated tokens.
// Heads are processed in pairs: one pass over x feeds both heads' QKV tiles,
// halving global x traffic and f32->f16 conversion work.
__global__ __launch_bounds__(256) void k_attn(
    const float* __restrict__ x1, const float* __restrict__ x2,
    const _Float16* __restrict__ wqkv, const float* __restrict__ qb,
    const float* __restrict__ vb, const float* __restrict__ ls,
    const float* __restrict__ biasb, const _Float16* __restrict__ wproj,
    const float* __restrict__ pb, float* __restrict__ out) {
  constexpr int C = 512;
  __shared__ __align__(32) _Float16 sOut[128 * 512];   // 128 KB: pre-proj out (f16)
  __shared__ __align__(32) float    sQKV[128 * 192];   // 96 KB: raw qkv f32 (2 heads); reused as P
  __shared__ __align__(32) _Float16 sQ[2 * 128 * 32];  // 16 KB  normalized q
  __shared__ __align__(32) _Float16 sK[2 * 128 * 32];  // 16 KB  normalized k [j][d]
  __shared__ __align__(32) _Float16 sVT[2 * 32 * 128]; // 16 KB  v transposed [d][row]

  const int b    = blockIdx.x;
  const int tid  = threadIdx.x;
  const int lane = tid & 31;
  const int w    = tid >> 5;       // wave id 0..7
  const int n    = lane & 15;      // N-column / A-row within tile
  const bool hi  = lane >= 16;
  const int mo   = hi ? 8 : 0;     // M offset for D-layout vgprs
  const int rb   = w * 16;         // this wave's row block

  // Global source row for this lane's A-tile row (rows 0..63 = x1, 64..127 = x2)
  const int rowg = rb + n;
  const float* xrow = (rowg < 64) ? (x1 + ((size_t)b * 64 + rowg) * C)
                                  : (x2 + ((size_t)b * 64 + (rowg - 64)) * C);

  for (int hh = 0; hh < 8; ++hh) {         // head pair
    // ---- Phase 1: qkv GEMM for heads (2hh, 2hh+1): [128x512]*[512x192] ----
    int wbase[12];
#pragma unroll
    for (int hs = 0; hs < 2; ++hs) {
      const int hb = (hh * 2 + hs) * 32;
      wbase[hs * 6 + 0] = hb;            wbase[hs * 6 + 1] = hb + 16;
      wbase[hs * 6 + 2] = 512 + hb;      wbase[hs * 6 + 3] = 512 + hb + 16;
      wbase[hs * 6 + 4] = 1024 + hb;     wbase[hs * 6 + 5] = 1024 + hb + 16;
    }
    v8f acc[12];
#pragma unroll
    for (int t = 0; t < 12; ++t) acc[t] = (v8f){};

    for (int kk = 0; kk < 512; kk += 32) {
      const int k0 = kk + (hi ? 16 : 0);
      float4 f0 = *(const float4*)(xrow + k0);
      float4 f1 = *(const float4*)(xrow + k0 + 4);
      float4 f2 = *(const float4*)(xrow + k0 + 8);
      float4 f3 = *(const float4*)(xrow + k0 + 12);
      v16h av;
      av[0] = (_Float16)f0.x;  av[1] = (_Float16)f0.y;
      av[2] = (_Float16)f0.z;  av[3] = (_Float16)f0.w;
      av[4] = (_Float16)f1.x;  av[5] = (_Float16)f1.y;
      av[6] = (_Float16)f1.z;  av[7] = (_Float16)f1.w;
      av[8] = (_Float16)f2.x;  av[9] = (_Float16)f2.y;
      av[10] = (_Float16)f2.z; av[11] = (_Float16)f2.w;
      av[12] = (_Float16)f3.x; av[13] = (_Float16)f3.y;
      av[14] = (_Float16)f3.z; av[15] = (_Float16)f3.w;
      const int kb = kk + (hi ? 16 : 0);
#pragma unroll
      for (int t = 0; t < 12; ++t) {
        v16h bt = *(const v16h*)(wqkv + (size_t)(wbase[t] + n) * 512 + kb);
        acc[t] = wmma16(av, bt, acc[t]);
      }
    }
#pragma unroll
    for (int t = 0; t < 12; ++t)
#pragma unroll
      for (int r = 0; r < 8; ++r)
        sQKV[(rb + r + mo) * 192 + t * 16 + n] = acc[t][r];
    __syncthreads();

    // ---- Phase 2: bias + cosine-normalize, stage for WMMA ----
    {
      const int i = tid & 127;
      const int hs = tid >> 7;                    // 0 or 1: which head of the pair
      const int head = hh * 2 + hs;
      const int isrc = (i < 64) ? i : (i - 64);   // k2 := k1 (reference quirk)
      const float* base = sQKV + i * 192 + hs * 96;
      const float* kbase = sQKV + isrc * 192 + hs * 96 + 32;
      float q[32], k[32];
      float sq = 0.f, sk = 0.f;
#pragma unroll
      for (int c = 0; c < 32; ++c) {
        q[c] = base[c] + qb[head * 32 + c];
        sq = fmaf(q[c], q[c], sq);
        k[c] = kbase[c];
        sk = fmaf(k[c], k[c], sk);
      }
      const float rq = 1.f / fmaxf(sqrtf(sq), 1e-12f);
      const float rk = 1.f / fmaxf(sqrtf(sk), 1e-12f);
      _Float16* dq = sQ + hs * 4096 + i * 32;
      _Float16* dk = sK + hs * 4096 + i * 32;
      _Float16* dv = sVT + hs * 4096;
#pragma unroll
      for (int c = 0; c < 32; ++c) {
        dq[c] = (_Float16)(q[c] * rq);
        dk[c] = (_Float16)(k[c] * rk);
        float vv = base[64 + c] + vb[head * 32 + c];
        dv[c * 128 + i] = (_Float16)vv;
      }
    }
    __syncthreads();

    // ---- Phase 3: QK^T, softmax, PV (per-wave 16-row tile, both heads) ----
    for (int hs = 0; hs < 2; ++hs) {
      const int h = hh * 2 + hs;
      const float scale = __expf(fminf(ls[h], 4.6051701859880914f)); // min(ls, ln100)
      v16h aq = load_a16(sQ + hs * 4096 + (rb + n) * 32 + (hi ? 16 : 0));
      float pv[8][8], rmax[8], rsum[8];
#pragma unroll
      for (int r = 0; r < 8; ++r) { rmax[r] = -3.0e38f; rsum[r] = 0.f; }
#pragma unroll
      for (int jt = 0; jt < 8; ++jt) {
        v16h bk = *(const v16h*)(sK + hs * 4096 + (jt * 16 + n) * 32 + (hi ? 16 : 0));
        v8f z = (v8f){};
        v8f at = wmma16(aq, bk, z);
#pragma unroll
        for (int r = 0; r < 8; ++r) {
          float val = at[r] * scale +
                      biasb[(h * 128 + (rb + r + mo)) * 128 + jt * 16 + n];
          pv[jt][r] = val;
          rmax[r] = fmaxf(rmax[r], val);
        }
      }
#pragma unroll
      for (int r = 0; r < 8; ++r) {
#pragma unroll
        for (int mk = 1; mk < 16; mk <<= 1)
          rmax[r] = fmaxf(rmax[r], __shfl_xor(rmax[r], mk, 32));
      }
      _Float16* sP = reinterpret_cast<_Float16*>(sQKV) + w * 2048; // 16x128 per wave
#pragma unroll
      for (int jt = 0; jt < 8; ++jt)
#pragma unroll
        for (int r = 0; r < 8; ++r) {
          float p = __expf(pv[jt][r] - rmax[r]);
          rsum[r] += p;
          sP[(r + mo) * 128 + jt * 16 + n] = (_Float16)p;
        }
#pragma unroll
      for (int r = 0; r < 8; ++r) {
#pragma unroll
        for (int mk = 1; mk < 16; mk <<= 1)
          rsum[r] += __shfl_xor(rsum[r], mk, 32);
        rsum[r] = 1.f / rsum[r];
      }
#pragma unroll
      for (int nt = 0; nt < 2; ++nt) {
        v8f o = (v8f){};
#pragma unroll
        for (int kt = 0; kt < 4; ++kt) {
          v16h ap = load_a16(sP + n * 128 + kt * 32 + (hi ? 16 : 0));
          v16h bv = *(const v16h*)(sVT + hs * 4096 +
                                   (nt * 16 + n) * 128 + kt * 32 + (hi ? 16 : 0));
          o = wmma16(ap, bv, o);
        }
#pragma unroll
        for (int r = 0; r < 8; ++r)
          sOut[(rb + r + mo) * 512 + h * 32 + nt * 16 + n] =
              (_Float16)(o[r] * rsum[r]);
      }
    }
    __syncthreads();
  } // head pairs

  // ---- Phase 4: output projection [128 x 512] * [512 x 512] + bias ----
  const size_t OUT1 = (size_t)512 * 64 * 512;
  for (int nc = 0; nc < 4; ++nc) {
    v8f acc[8];
#pragma unroll
    for (int t = 0; t < 8; ++t) acc[t] = (v8f){};
    for (int kk = 0; kk < 512; kk += 32) {
      v16h a = load_a16(sOut + (rb + n) * 512 + kk + (hi ? 16 : 0));
      const int kb = kk + (hi ? 16 : 0);
#pragma unroll
      for (int t = 0; t < 8; ++t) {
        v16h bw = *(const v16h*)(wproj + (size_t)(nc * 128 + t * 16 + n) * 512 + kb);
        acc[t] = wmma16(a, bw, acc[t]);
      }
    }
#pragma unroll
    for (int t = 0; t < 8; ++t)
#pragma unroll
      for (int r = 0; r < 8; ++r) {
        const int c = nc * 128 + t * 16 + n;
        const int i = rb + r + mo;
        const float val = acc[t][r] + pb[c];
        const size_t off = (i < 64)
            ? (((size_t)b * 64 + i) * 512 + c)
            : (OUT1 + ((size_t)b * 64 + (i - 64)) * 512 + c);
        out[off] = val;
      }
  }
}

// ---------------------------------------------------------------------------
extern "C" void kernel_launch(void* const* d_in, const int* in_sizes, int n_in,
                              void* d_out, int out_size, void* d_ws, size_t ws_size,
                              hipStream_t stream) {
  (void)in_sizes; (void)n_in; (void)out_size; (void)ws_size;
  const float* x1     = (const float*)d_in[0];
  const float* x2     = (const float*)d_in[1];
  const float* qkv_w  = (const float*)d_in[2];
  const float* q_bias = (const float*)d_in[3];
  const float* v_bias = (const float*)d_in[4];
  const float* logit  = (const float*)d_in[5];
  const float* cpb_w1 = (const float*)d_in[6];
  const float* cpb_b1 = (const float*)d_in[7];
  const float* cpb_w2 = (const float*)d_in[8];
  const float* a1     = (const float*)d_in[9];
  const float* a2     = (const float*)d_in[10];
  const float* proj_w = (const float*)d_in[11];
  const float* proj_b = (const float*)d_in[12];
  const float* rct    = (const float*)d_in[13];
  const int*   rpi    = (const int*)d_in[14];
  float* out = (float*)d_out;

  char* wsp = (char*)d_ws;
  _Float16* wqkv  = (_Float16*)wsp; wsp += (size_t)786432 * 2;
  _Float16* wproj = (_Float16*)wsp; wsp += (size_t)262144 * 2;
  float* biasbuf  = (float*)wsp;    wsp += (size_t)262144 * 4;
  float* table    = (float*)wsp;    wsp += (size_t)3600 * 4;

  k_cvt  <<<3072, 256, 0, stream>>>(qkv_w, proj_w, wqkv, wproj);
  k_table<<<15,   256, 0, stream>>>(rct, cpb_w1, cpb_b1, cpb_w2, table);
  k_bias <<<1024, 256, 0, stream>>>(rpi, table, a1, a2, biasbuf);
  k_attn <<<512,  256, 0, stream>>>(x1, x2, wqkv, q_bias, v_bias, logit,
                                    biasbuf, wproj, proj_b, out);
}
